// VectorAttentionLayerV3_28432683500128
// MI455X (gfx1250) — matrix operand order repaired and checked by reading, not compiled, not run on hardware
//
#include <hip/hip_runtime.h>
#include <math.h>

typedef __attribute__((ext_vector_type(16))) __bf16 v16bf;
typedef __attribute__((ext_vector_type(8)))  float  v8f;

#define LDS_FENCE() asm volatile("" ::: "memory")

__device__ __forceinline__ unsigned int pk_bf16(float a, float b) {
  unsigned int ua = __builtin_bit_cast(unsigned int, a);
  unsigned int ub = __builtin_bit_cast(unsigned int, b);
  ua = (ua + 0x7FFFu + ((ua >> 16) & 1u)) >> 16;
  ub = (ub + 0x7FFFu + ((ub >> 16) & 1u)) >> 16;
  return ua | (ub << 16);
}
__device__ __forceinline__ unsigned short bf16_1(float a) {
  unsigned int ua = __builtin_bit_cast(unsigned int, a);
  return (unsigned short)((ua + 0x7FFFu + ((ua >> 16) & 1u)) >> 16);
}

union V16U { v16bf v; uint4 u[2]; unsigned int s[8]; };

// A-operand (16x32 bf16 MxK) from row-major [16][64] *bf16* LDS tile.
__device__ __forceinline__ v16bf loadA(const unsigned short* buf, int kt, int lane) {
  int m  = lane & 15;
  int k0 = kt * 32 + ((lane & 16) ? 8 : 0);
  V16U r;
  r.u[0] = *(const uint4*)(buf + m * 64 + k0);
  r.u[1] = *(const uint4*)(buf + m * 64 + k0 + 16);
  return r.v;
}
// A-operand from row-major [16][64] *fp32* LDS tile (async-DMA'd), cvt on read.
__device__ __forceinline__ v16bf loadA_cvt(const float* buf, int kt, int lane) {
  int m  = lane & 15;
  int k0 = kt * 32 + ((lane & 16) ? 8 : 0);
  const float* p = buf + m * 64 + k0;
  float f[16];
#pragma unroll
  for (int j = 0; j < 8; ++j) f[j] = p[j];
#pragma unroll
  for (int j = 0; j < 8; ++j) f[8 + j] = p[16 + j];
  V16U r;
#pragma unroll
  for (int j = 0; j < 8; ++j) r.s[j] = pk_bf16(f[2 * j], f[2 * j + 1]);
  return r.v;
}
// B-operand (32x16 bf16 KxN) from row-major [out][64] bf16 weight LDS.
__device__ __forceinline__ v16bf loadB(const unsigned short* w, int nt, int kt, int lane) {
  int n  = nt * 16 + (lane & 15);
  int k0 = kt * 32 + ((lane & 16) ? 8 : 0);
  V16U r;
  r.u[0] = *(const uint4*)(w + n * 64 + k0);
  r.u[1] = *(const uint4*)(w + n * 64 + k0 + 16);
  return r.v;
}

__device__ __forceinline__ v8f wmma_bf16(v16bf a, v16bf b, v8f c) {
  return __builtin_amdgcn_wmma_f32_16x16x32_bf16(false, a, false, b, (short)0, c,
                                                 false, false);
}

// Async DMA one 16x64 fp32 tile (4 KB) global -> LDS; 8 x b128 per lane.
__device__ __forceinline__ void async_tile(const float* __restrict__ g, float* l,
                                           int lane) {
#pragma unroll
  for (int ch = 0; ch < 8; ++ch) {
    int e = ch * 128 + lane * 4;
    unsigned loff = (unsigned)(size_t)(l + e);  // generic->LDS byte offset
    asm volatile("global_load_async_to_lds_b128 %0, %1, off"
                 :: "v"(loff), "v"(g + e) : "memory");
  }
}

// Cooperative 16x64 fp32 global tile -> bf16 row-major LDS staging (per wave).
__device__ __forceinline__ void stage_tile_bf16(const float* __restrict__ src,
                                                unsigned short* stg, int lane) {
#pragma unroll
  for (int ch = 0; ch < 8; ++ch) {
    int e = ch * 128 + lane * 4;
    const float4 f = *(const float4*)(src + e);
    uint2 pkd;
    pkd.x = pk_bf16(f.x, f.y);
    pkd.y = pk_bf16(f.z, f.w);
    *(uint2*)(stg + e) = pkd;
  }
}

__global__ void __launch_bounds__(128) vattn_kernel(
    const float* __restrict__ center, const float* __restrict__ neighbor,
    const float* __restrict__ pos,    const float* __restrict__ Wq,
    const float* __restrict__ Wkv,    const float* __restrict__ W1,
    const float* __restrict__ gamma,  const float* __restrict__ beta,
    const float* __restrict__ mean,   const float* __restrict__ var,
    const float* __restrict__ W2,     float* __restrict__ out) {
  extern __shared__ unsigned char smem[];
  unsigned short* wq  = (unsigned short*)smem;   // 64x64
  unsigned short* wkv = wq  + 4096;              // 128x64
  unsigned short* w1  = wkv + 8192;              // 64x64
  unsigned short* w2  = w1  + 4096;              // 64x64  (weights: 40 KB)

  const int tid  = threadIdx.x;
  const int wave = tid >> 5;
  const int lane = tid & 31;
  unsigned char* pw = smem + 40960 + wave * 22528;
  unsigned short* stg  = (unsigned short*)pw;     // 16x64 bf16 staging (2 KB)
  float* qbuf    = (float*)(pw + 2048);           // 16x64 f32 Q        (4 KB)
  float* nfbuf[2]  = {(float*)(pw + 6144),  (float*)(pw + 10240)};  // 2x4 KB
  float* posbuf[2] = {(float*)(pw + 14336), (float*)(pw + 18432)};  // 2x4 KB

  const int g  = blockIdx.x * 4 + wave;  // 16-point group id (2048 total)
  const int p0 = g * 16;                 // flat point index = b*N + n

  // Kick off async DMA of point 0's tiles immediately (overlaps setup + Q GEMM).
  async_tile(neighbor + (size_t)p0 * 1024, nfbuf[0], lane);
  async_tile(pos      + (size_t)p0 * 1024, posbuf[0], lane);

  // ---- weights -> bf16 LDS (row-major [out][in]) ----
  for (int i = tid; i < 4096; i += 128) wq[i]  = bf16_1(Wq[i]);
  for (int i = tid; i < 8192; i += 128) wkv[i] = bf16_1(Wkv[i]);
  for (int i = tid; i < 4096; i += 128) w1[i]  = bf16_1(W1[i]);
  for (int i = tid; i < 4096; i += 128) w2[i]  = bf16_1(W2[i]);
  __syncthreads();

  const int cc = lane & 15;
  const int mb = (lane & 16) ? 8 : 0;

  // ---- per-lane BN scale/shift for channels c = t*16 + cc ----
  float bns[4], bnb[4];
#pragma unroll
  for (int t = 0; t < 4; ++t) {
    int ch = t * 16 + cc;
    float s = gamma[ch] * __frsqrt_rn(var[ch] + 1e-5f);
    bns[t]  = s;
    bnb[t]  = beta[ch] - mean[ch] * s;
  }

  // ---- Q for this wave's 16 points: 16x64 = center @ Wq^T ----
  {
    stage_tile_bf16(center + (size_t)p0 * 64, stg, lane);
    LDS_FENCE();
    v16bf a0 = loadA(stg, 0, lane), a1 = loadA(stg, 1, lane);
#pragma unroll
    for (int t = 0; t < 4; ++t) {
      v8f acc = {};
      acc = wmma_bf16(a0, loadB(wq, t, 0, lane), acc);
      acc = wmma_bf16(a1, loadB(wq, t, 1, lane), acc);
#pragma unroll
      for (int r = 0; r < 8; ++r) qbuf[(mb + r) * 64 + t * 16 + cc] = acc[r];
    }
    LDS_FENCE();
  }

#pragma unroll 1
  for (int i = 0; i < 16; ++i) {
    const float* nfb = nfbuf[i & 1];
    const float* psb = posbuf[i & 1];

    // Prefetch next point's tiles into the other buffer, then wait for current.
    if (i < 15) {
      size_t pbn = (size_t)(p0 + i + 1) * 1024;
      async_tile(neighbor + pbn, nfbuf[(i + 1) & 1], lane);
      async_tile(pos + pbn,      posbuf[(i + 1) & 1], lane);
      asm volatile("s_wait_asynccnt 0x10" ::: "memory");  // only next 16 pending
    } else {
      asm volatile("s_wait_asynccnt 0x0" ::: "memory");
    }

    // pos_enc in WMMA C/D register layout (from LDS)
    v8f posv[4];
#pragma unroll
    for (int t = 0; t < 4; ++t)
#pragma unroll
      for (int r = 0; r < 8; ++r)
        posv[t][r] = psb[(mb + r) * 64 + t * 16 + cc];

    // A-operands from fp32 LDS tile, cvt->bf16 on read
    v16bf a0 = loadA_cvt(nfb, 0, lane), a1 = loadA_cvt(nfb, 1, lane);

    // K,V = nf @ Wkv^T : 8 col tiles (0..3 = K, 4..7 = V)
    v8f kac[4], vac[4];
#pragma unroll
    for (int t = 0; t < 8; ++t) {
      v8f acc = {};
      acc = wmma_bf16(a0, loadB(wkv, t, 0, lane), acc);
      acc = wmma_bf16(a1, loadB(wkv, t, 1, lane), acc);
      if (t < 4) kac[t] = acc; else vac[t - 4] = acc;
    }

    // attn_pre = Q[i] - K + pos  -> bf16 staging (row-major)
#pragma unroll
    for (int t = 0; t < 4; ++t) {
      float q = qbuf[i * 64 + t * 16 + cc];
#pragma unroll
      for (int r = 0; r < 8; ++r)
        stg[(mb + r) * 64 + t * 16 + cc] = bf16_1(q - kac[t][r] + posv[t][r]);
    }
    LDS_FENCE();
    a0 = loadA(stg, 0, lane);
    a1 = loadA(stg, 1, lane);
    LDS_FENCE();

    // h = relu(BN(attn @ W1^T)) -> bf16 staging
#pragma unroll
    for (int t = 0; t < 4; ++t) {
      v8f acc = {};
      acc = wmma_bf16(a0, loadB(w1, t, 0, lane), acc);
      acc = wmma_bf16(a1, loadB(w1, t, 1, lane), acc);
#pragma unroll
      for (int r = 0; r < 8; ++r) {
        float h = fmaxf(acc[r] * bns[t] + bnb[t], 0.0f);
        stg[(mb + r) * 64 + t * 16 + cc] = bf16_1(h);
      }
    }
    LDS_FENCE();
    a0 = loadA(stg, 0, lane);
    a1 = loadA(stg, 1, lane);
    LDS_FENCE();

    // attn2 = h @ W2^T ; softmax over M (row dim) ; out = sum_m w*(V+pos)
    float* op = out + (size_t)(p0 + i) * 64;
#pragma unroll
    for (int t = 0; t < 4; ++t) {
      v8f acc = {};
      acc = wmma_bf16(a0, loadB(w2, t, 0, lane), acc);
      acc = wmma_bf16(a1, loadB(w2, t, 1, lane), acc);

      float mx = acc[0];
#pragma unroll
      for (int r = 1; r < 8; ++r) mx = fmaxf(mx, acc[r]);
      mx = fmaxf(mx, __shfl_xor(mx, 16, 32));

      float e_[8];
      float s = 0.0f;
#pragma unroll
      for (int r = 0; r < 8; ++r) {
        e_[r] = __expf(acc[r] - mx);
        s += e_[r];
      }
      s += __shfl_xor(s, 16, 32);
      float inv = 1.0f / s;

      float o = 0.0f;
#pragma unroll
      for (int r = 0; r < 8; ++r) o += e_[r] * (vac[t][r] + posv[t][r]);
      o *= inv;
      o += __shfl_xor(o, 16, 32);
      if (lane < 16) op[t * 16 + cc] = o;
    }
  }
}

extern "C" void kernel_launch(void* const* d_in, const int* in_sizes, int n_in,
                              void* d_out, int out_size, void* d_ws, size_t ws_size,
                              hipStream_t stream) {
  (void)in_sizes; (void)n_in; (void)out_size; (void)d_ws; (void)ws_size;
  const float* center   = (const float*)d_in[0];
  const float* neighbor = (const float*)d_in[1];
  const float* pos      = (const float*)d_in[2];
  const float* Wq       = (const float*)d_in[3];
  const float* Wkv      = (const float*)d_in[4];
  const float* W1       = (const float*)d_in[5];
  const float* gamma    = (const float*)d_in[6];
  const float* beta     = (const float*)d_in[7];
  const float* mean     = (const float*)d_in[8];
  const float* var      = (const float*)d_in[9];
  const float* W2       = (const float*)d_in[10];
  float* out            = (float*)d_out;

  // B*N = 32768 points -> 2048 groups of 16 -> 4 waves/block -> 512 blocks.
  dim3 grid(512), block(128);
  // weights 40 KB + 4 waves * (2 KB stg + 4 KB qbuf + 2*4 KB nf + 2*4 KB pos)
  size_t lds_bytes = 40960 + 4 * 22528;  // 128 KB
  hipLaunchKernelGGL(vattn_kernel, grid, block, lds_bytes, stream,
                     center, neighbor, pos, Wq, Wkv, W1, gamma, beta, mean, var,
                     W2, out);
}
